// Attention_54236847014114
// MI455X (gfx1250) — matrix operand order repaired and checked by reading, not compiled, not run on hardware
//
#include <hip/hip_runtime.h>
#include <hip/hip_bf16.h>

// ---------------------------------------------------------------------------
// MHA forward for MI455X (gfx1250), wave32 + v_wmma_f32_16x16x32_bf16.
// - Fused scores+softmax+AV kernel keeps the 16x2048 score strip in LDS
//   (128 KB -- enabled by CDNA5's 320 KB LDS); only HBM traffic for the
//   attention middle section is the mandatory 536 MB attn write.
// - attn is streamed with non-temporal stores so the 536 MB stream does not
//   evict the L2-resident Q/K/V/weight working set (~40 MB in 192 MB L2).
// - Projection GEMMs use 32x64 output blocks per wave for 2x B-operand reuse.
// ---------------------------------------------------------------------------

typedef __attribute__((ext_vector_type(16))) __bf16 v16bf;
typedef __attribute__((ext_vector_type(8)))  float  v8f;
typedef __attribute__((ext_vector_type(4)))  float  v4f;   // native vector for NT stores

#define WMMA_BF16(a, b, c) \
  __builtin_amdgcn_wmma_f32_16x16x32_bf16(false, (a), false, (b), (short)0, (c), false, false)

constexpr int  Bb = 2, Nn = 2048, Cc = 1024, Hh = 16, Dd = 64;
constexpr float SCALE = 1.0f / 32.0f;  // 1/sqrt(C) = 1/sqrt(1024)
constexpr int  SP = 2048 + 4;          // LDS row stride (floats): +16B to spread banks

// ---- A-tile loader: 16x32 bf16 A matrix --------------------------------
// lane = M row (lane&15); half = lane>>4.
// K runs: j in [0,8) -> K = 8*half + j ; j in [8,16) -> K = 16 + 8*half + (j-8)
__device__ __forceinline__ v16bf load_a_f32(const float* __restrict__ p, int ld) {
  const int lane = threadIdx.x & 31;
  const int row  = lane & 15;
  const int half = lane >> 4;
  const float* r0 = p + (size_t)row * ld + half * 8;
  const float* r1 = r0 + 16;
  v16bf t;
#pragma unroll
  for (int j = 0; j < 8; ++j) { t[j] = (__bf16)r0[j]; t[j + 8] = (__bf16)r1[j]; }
  return t;
}

__device__ __forceinline__ v16bf load_a_bf16(const __bf16* __restrict__ p, int ld) {
  const int lane = threadIdx.x & 31;
  const int row  = lane & 15;
  const int half = lane >> 4;
  const __bf16* r0 = p + (size_t)row * ld + half * 8;
  const __bf16* r1 = r0 + 16;
  v16bf t;
#pragma unroll
  for (int j = 0; j < 8; ++j) { t[j] = r0[j]; t[j + 8] = r1[j]; }
  return t;
}

// A-tile from LDS-resident f32 strip (rows are the 16 query rows).
__device__ __forceinline__ v16bf load_a_lds(const float* p, int ld) {
  const int lane = threadIdx.x & 31;
  const int row  = lane & 15;
  const int half = lane >> 4;
  const float* r0 = p + row * ld + half * 8;
  const float* r1 = r0 + 16;
  v16bf t;
#pragma unroll
  for (int j = 0; j < 8; ++j) { t[j] = (__bf16)r0[j]; t[j + 8] = (__bf16)r1[j]; }
  return t;
}

// ---- B-tile loader: 32x16 bf16 B matrix --------------------------------
// lane = N column (lane&15); half = lane>>4: j<8 -> K = 8*half + j ; j>=8 -> K = 16 + 8*half + (j-8).
// Source memory layout: row-major with one memory row per B *column*.
__device__ __forceinline__ v16bf load_b_f32(const float* __restrict__ p, int ld) {
  const int lane = threadIdx.x & 31;
  const float* r = p + (size_t)(lane & 15) * ld + ((lane >> 4) * 8);
  v16bf t;
#pragma unroll
  for (int j = 0; j < 8; ++j) { t[j] = (__bf16)r[j]; t[j + 8] = (__bf16)r[j + 16]; }
  return t;
}

__device__ __forceinline__ v16bf load_b_bf16(const __bf16* __restrict__ p, int ld) {
  const int lane = threadIdx.x & 31;
  const __bf16* r = p + (size_t)(lane & 15) * ld + ((lane >> 4) * 8);
  v16bf t;
#pragma unroll
  for (int j = 0; j < 8; ++j) { t[j] = r[j]; t[j + 8] = r[j + 16]; }
  return t;
}

// ---------------------------------------------------------------------------
// Kernel 1: qkv = x @ w_qkv^T ; scatter Q,K as [B,H,N,D] bf16, V as [B,H,D,N].
// One wave -> 32 rows x 64 cols (8 accumulators, 2x B reuse).
// Jobs = 128 row-tiles * 48 col-groups = 6144 waves -> 768 blocks of 8 waves.
// ---------------------------------------------------------------------------
__global__ __launch_bounds__(256) void k_qkv(const float* __restrict__ x,
                                             const float* __restrict__ w,
                                             __bf16* __restrict__ qb,
                                             __bf16* __restrict__ kb,
                                             __bf16* __restrict__ vt) {
  const int wid = blockIdx.x * (blockDim.x >> 5) + (threadIdx.x >> 5);
  const int m0  = (wid / 48) * 32;   // row tile over B*N = 4096
  const int c0  = (wid % 48) * 64;   // col group over 3C = 3072

  v8f acc[2][4] = {{v8f{0}, v8f{0}, v8f{0}, v8f{0}},
                   {v8f{0}, v8f{0}, v8f{0}, v8f{0}}};
  for (int kk = 0; kk < Cc; kk += 32) {
    v16bf a0 = load_a_f32(x + (size_t)m0 * Cc + kk, Cc);
    v16bf a1 = load_a_f32(x + (size_t)(m0 + 16) * Cc + kk, Cc);
#pragma unroll
    for (int j = 0; j < 4; ++j) {
      v16bf bt = load_b_f32(w + (size_t)(c0 + 16 * j) * Cc + kk, Cc);
      acc[0][j] = WMMA_BF16(a0, bt, acc[0][j]);
      acc[1][j] = WMMA_BF16(a1, bt, acc[1][j]);
    }
  }

  const int lane = threadIdx.x & 31;
  const int col16 = lane & 15, half = lane >> 4;
  const int which = c0 / Cc;        // 0=Q, 1=K, 2=V
  const int h     = (c0 % Cc) / Dd; // 64-col group == one full head dim
#pragma unroll
  for (int t = 0; t < 2; ++t)
#pragma unroll
    for (int j = 0; j < 4; ++j) {
      const int d = 16 * j + col16;
#pragma unroll
      for (int r = 0; r < 8; ++r) {
        const int m = m0 + 16 * t + r + 8 * half;  // global row over B*N
        const int b = m >> 11, n = m & 2047;
        const __bf16 v = (__bf16)acc[t][j][r];
        const size_t hb = (size_t)(b * Hh + h);
        if (which == 0)      qb[(hb * Nn + n) * Dd + d] = v;
        else if (which == 1) kb[(hb * Nn + n) * Dd + d] = v;
        else                 vt[(hb * Dd + d) * Nn + n] = v;  // V transposed
      }
    }
}

// ---------------------------------------------------------------------------
// Kernel 2 (fused): S = QK^T*scale -> LDS ; softmax in LDS -> attn (f32, NT) ;
// O = P@V from LDS -> ob (bf16).
// One workgroup = 4 waves = 128 threads handles (head bh, 16 query rows).
// Grid = 32 heads * 128 row-blocks = 4096 workgroups.
// ---------------------------------------------------------------------------
__global__ __launch_bounds__(128) void k_attn(const __bf16* __restrict__ qb,
                                              const __bf16* __restrict__ kb,
                                              const __bf16* __restrict__ vt,
                                              float* __restrict__ attn,
                                              __bf16* __restrict__ ob) {
  __shared__ float S[16 * SP];      // 16 x 2048 f32 score strip (+pad) ~128.25 KB
  __shared__ float red[128];
  __shared__ float rowinv[16];

  const int bh = blockIdx.x >> 7;            // head index over B*H = 32
  const int n0 = (blockIdx.x & 127) * 16;    // query row block
  const int wv = threadIdx.x >> 5;           // wave 0..3
  const int lane = threadIdx.x & 31;
  const int col16 = lane & 15, half = lane >> 4;

  const __bf16* qh = qb + (size_t)bh * Nn * Dd;
  const __bf16* kh = kb + (size_t)bh * Nn * Dd;
  const __bf16* vh = vt + (size_t)bh * Dd * Nn;

  // ---- Phase 1: scores. Wave wv computes columns [wv*512, wv*512+512). ----
  v16bf a0 = load_a_bf16(qh + (size_t)n0 * Dd + 0,  Dd);
  v16bf a1 = load_a_bf16(qh + (size_t)n0 * Dd + 32, Dd);
  for (int g = 0; g < 8; ++g) {
    const int mbase = wv * 512 + g * 64;
    // prefetch next group's K rows (L2 warm-up)
    if (g < 7)
      __builtin_prefetch((const char*)(kh + (size_t)(mbase + 64 + (lane & 15) * 4) * Dd), 0, 0);
    v8f acc[4] = {v8f{0}, v8f{0}, v8f{0}, v8f{0}};
#pragma unroll
    for (int j = 0; j < 4; ++j) {
      v16bf b0 = load_b_bf16(kh + (size_t)(mbase + 16 * j) * Dd + 0,  Dd);
      acc[j] = WMMA_BF16(a0, b0, acc[j]);
      v16bf b1 = load_b_bf16(kh + (size_t)(mbase + 16 * j) * Dd + 32, Dd);
      acc[j] = WMMA_BF16(a1, b1, acc[j]);
    }
#pragma unroll
    for (int j = 0; j < 4; ++j)
#pragma unroll
      for (int r = 0; r < 8; ++r)
        S[(r + 8 * half) * SP + mbase + 16 * j + col16] = acc[j][r] * SCALE;
  }
  __syncthreads();

  // ---- Phase 2: softmax. 8 threads per row; each owns 256 columns. ----
  const int row = threadIdx.x >> 3;   // 0..15
  const int cid = threadIdx.x & 7;    // 0..7
  float* Sr = S + row * SP + cid * 256;

  float mx = -3.0e38f;
#pragma unroll 4
  for (int i = 0; i < 256; ++i) mx = fmaxf(mx, Sr[i]);
  red[threadIdx.x] = mx;
  __syncthreads();
  if (cid == 0) {
    float m = red[row * 8];
#pragma unroll
    for (int i = 1; i < 8; ++i) m = fmaxf(m, red[row * 8 + i]);
    rowinv[row] = m;                  // temporarily holds row max
  }
  __syncthreads();
  mx = rowinv[row];

  float sum = 0.0f;
#pragma unroll 4
  for (int i = 0; i < 256; ++i) {
    float e = __expf(Sr[i] - mx);
    Sr[i] = e;
    sum += e;
  }
  red[threadIdx.x] = sum;
  __syncthreads();
  if (cid == 0) {
    float s = red[row * 8];
#pragma unroll
    for (int i = 1; i < 8; ++i) s += red[row * 8 + i];
    rowinv[row] = 1.0f / s;
  }
  __syncthreads();
  const float inv = rowinv[row];

  // Normalize in LDS and stream attn row block to global with NT stores
  // (written once, never re-read from global -> keep it out of L2).
  float* gout = attn + (size_t)bh * Nn * Nn + (size_t)(n0 + row) * Nn + cid * 256;
#pragma unroll 2
  for (int i = 0; i < 256; i += 4) {
    v4f v4;
    v4.x = Sr[i + 0] * inv;  v4.y = Sr[i + 1] * inv;
    v4.z = Sr[i + 2] * inv;  v4.w = Sr[i + 3] * inv;
    Sr[i + 0] = v4.x; Sr[i + 1] = v4.y; Sr[i + 2] = v4.z; Sr[i + 3] = v4.w;
    __builtin_nontemporal_store(v4, (v4f*)(gout + i));
  }
  // Prefetch V^T block for phase 3 while stores drain.
  {
    const char* vbase = (const char*)vh;
    for (int i = 0; i < 8; ++i)
      __builtin_prefetch(vbase + ((size_t)threadIdx.x * 8 + i) * 256, 0, 0);
  }
  __syncthreads();

  // ---- Phase 3: O = P @ V. Wave wv owns d-tile [16*wv, 16*wv+16). ----
  v8f acc = v8f{0};
  for (int mm = 0; mm < Nn; mm += 32) {
    v16bf a  = load_a_lds(S + mm, SP);
    v16bf bt = load_b_bf16(vh + (size_t)(16 * wv) * Nn + mm, Nn);
    acc = WMMA_BF16(a, bt, acc);
  }
  const int b = bh >> 4, h = bh & 15;
#pragma unroll
  for (int r = 0; r < 8; ++r) {
    const int n = n0 + r + 8 * half;
    ob[(size_t)(b * Nn + n) * Cc + h * Dd + 16 * wv + col16] = (__bf16)acc[r];
  }
}

// ---------------------------------------------------------------------------
// Kernel 3: out = O @ w_proj^T + b_proj -> f32 out region of d_out.
// One wave -> 32 rows x 64 cols. Jobs = 128 * 16 = 2048 waves -> 256 blocks.
// ---------------------------------------------------------------------------
__global__ __launch_bounds__(256) void k_proj(const __bf16* __restrict__ ob,
                                              const float* __restrict__ w,
                                              const float* __restrict__ bias,
                                              float* __restrict__ out) {
  const int wid = blockIdx.x * (blockDim.x >> 5) + (threadIdx.x >> 5);
  const int m0  = (wid / 16) * 32;
  const int e0  = (wid % 16) * 64;

  v8f acc[2][4] = {{v8f{0}, v8f{0}, v8f{0}, v8f{0}},
                   {v8f{0}, v8f{0}, v8f{0}, v8f{0}}};
  for (int kk = 0; kk < Cc; kk += 32) {
    v16bf a0 = load_a_bf16(ob + (size_t)m0 * Cc + kk, Cc);
    v16bf a1 = load_a_bf16(ob + (size_t)(m0 + 16) * Cc + kk, Cc);
#pragma unroll
    for (int j = 0; j < 4; ++j) {
      v16bf bt = load_b_f32(w + (size_t)(e0 + 16 * j) * Cc + kk, Cc);
      acc[0][j] = WMMA_BF16(a0, bt, acc[0][j]);
      acc[1][j] = WMMA_BF16(a1, bt, acc[1][j]);
    }
  }

  const int lane = threadIdx.x & 31;
  const int col16 = lane & 15, half = lane >> 4;
#pragma unroll
  for (int t = 0; t < 2; ++t)
#pragma unroll
    for (int j = 0; j < 4; ++j) {
      const int e = e0 + 16 * j + col16;
      const float bsv = bias[e];
#pragma unroll
      for (int r = 0; r < 8; ++r)
        out[(size_t)(m0 + 16 * t + r + 8 * half) * Cc + e] = acc[t][j][r] + bsv;
    }
}

// ---------------------------------------------------------------------------
extern "C" void kernel_launch(void* const* d_in, const int* in_sizes, int n_in,
                              void* d_out, int out_size, void* d_ws, size_t ws_size,
                              hipStream_t stream) {
  const float* x      = (const float*)d_in[0];  // [B,N,C]
  const float* w_qkv  = (const float*)d_in[1];  // [3C,C]
  const float* w_proj = (const float*)d_in[2];  // [C,C]
  const float* b_proj = (const float*)d_in[3];  // [C]

  float* out  = (float*)d_out;                        // [B,N,C]
  float* attn = out + (size_t)Bb * Nn * Cc;           // [B,H,N,N]

  const size_t headElems = (size_t)Bb * Hh * Nn * Dd; // 4 Mi elems each
  __bf16* qb = (__bf16*)d_ws;
  __bf16* kb = qb + headElems;
  __bf16* vt = kb + headElems;                        // V transposed [B,H,D,N]
  __bf16* ob = vt + headElems;                        // O bf16 [B,N,C]

  k_qkv <<<768,  256, 0, stream>>>(x, w_qkv, qb, kb, vt);
  k_attn<<<4096, 128, 0, stream>>>(qb, kb, vt, attn, ob);
  k_proj<<<256,  256, 0, stream>>>(ob, w_proj, b_proj, out);
}
